// FPN_AAR_89163521065376
// MI455X (gfx1250) — matrix operand order repaired and checked by reading, not compiled, not run on hardware
//
#include <hip/hip_runtime.h>

typedef __attribute__((ext_vector_type(16))) _Float16 v16h;
typedef __attribute__((ext_vector_type(8)))  _Float16 v8h;
typedef __attribute__((ext_vector_type(8)))  float    v8f;
typedef __attribute__((ext_vector_type(4)))  int      v4i;

#define C_IN   256
#define C_OUT  256
#define HIN    48
#define WIN    48
#define HS     144          // sampled grid h*3
#define XH     146          // padded sampled height (covers oh0+lh+kh <= 145)
#define XW     160          // padded sampled width
#define HOUT   142
#define WOUT   142
#define NPIX   (HOUT*WOUT)  // 20164
#define KTOT   (C_IN*9)     // 2304
#define BATCH  2
#define BN_EPS 1e-5f
#define S2 1.41421356237309505f

// Async global->LDS staging (gfx1250, tracked by ASYNCcnt). Guarded so the
// file still compiles (with a synchronous fallback) if the builtin names
// differ on this toolchain. Signature (probe-confirmed via diagnostic):
//   (v4i addrspace(1)*, v4i addrspace(3)*, imm offset, imm cpol)
#if defined(__HIP_DEVICE_COMPILE__) && \
    __has_builtin(__builtin_amdgcn_global_load_async_to_lds_b128) && \
    __has_builtin(__builtin_amdgcn_s_wait_asynccnt)
#define USE_ASYNC 1
#else
#define USE_ASYNC 0
#endif

// Integer-cast route to address-space pointers: generic-LDS low 32 bits == LDS
// offset; generic-global value == global address.
#define ASG(p) ((__attribute__((address_space(1))) v4i*)(unsigned long long)(p))
#define ASL(p) ((__attribute__((address_space(3))) v4i*)(unsigned)(unsigned long long)(p))

__constant__ float c_ox[5][9] = {
  {0,0,0,0,0,0,0,0,0},
  {1.f-S2, 1.f-S2*0.5f, 1.f, -S2*0.5f, 0.f, S2*0.5f, -1.f, S2*0.5f-1.f, S2-1.f},
  {0,1,2,-1,0,1,-2,-1,0},
  {1.f, 1.f+S2*0.5f, 1.f+S2, -S2*0.5f, 0.f, S2*0.5f, -1.f-S2, -1.f-S2*0.5f, -1.f},
  {2,2,2,0,0,0,-2,-2,-2}
};
__constant__ float c_oy[5][9] = {
  {0,0,0,0,0,0,0,0,0},
  {1.f, S2*0.5f, S2-1.f, 1.f-S2*0.5f, 0.f, S2*0.5f-1.f, 1.f-S2, -S2*0.5f, -1.f},
  {2,1,0,1,0,-1,0,-1,-2},
  {1.f+S2, S2*0.5f, -1.f, 1.f+S2*0.5f, 0.f, -1.f-S2*0.5f, 1.f, -S2*0.5f, 1.f+S2},
  {2,0,-2,2,0,-2,2,0,-2}
};
__constant__ float c_pnx[9] = {-1,-1,-1,0,0,0,1,1,1};
__constant__ float c_pny[9] = {-1,0,1,-1,0,1,-1,0,1};

// ---------------- weights fp32 -> fp16 (A matrix, row = co, col = ci*9+kh*3+kw) ----
__global__ __launch_bounds__(256) void convert_w_kernel(const float* __restrict__ w,
                                                        _Float16* __restrict__ wh, int n) {
  int idx = blockIdx.x * 256 + threadIdx.x;
  if (idx < n) wh[idx] = (_Float16)w[idx];
}

// ---------------- separable bilinear sampling into padded f16 canvas ---------------
__global__ __launch_bounds__(256) void sample_kernel(const float* __restrict__ x,
                                                     _Float16* __restrict__ xo, int angle) {
  long long idx = (long long)blockIdx.x * 256 + threadIdx.x;
  const long long total = (long long)BATCH * C_IN * XH * XW;
  if (idx >= total) return;
  int pos = (int)(idx % (XH * XW));
  int bc  = (int)(idx / (XH * XW));
  int hh = pos / XW, ww = pos % XW;
  float val = 0.f;
  if (hh < HS && ww < HS) {
    int i = hh / 3, k1 = hh - i * 3;
    int j = ww / 3, k2 = ww - j * 3;
    int p = k1 * 3 + k2;
    float px = (float)(i + 1) + c_pnx[p] + c_ox[angle][p];
    float py = (float)(j + 1) + c_pny[p] + c_oy[angle][p];
    float flx = floorf(px), fly = floorf(py);
    float ltx = fminf(fmaxf(flx,       0.f), 49.f);
    float rbx = fminf(fmaxf(flx + 1.f, 0.f), 49.f);
    float pxc = fminf(fmaxf(px,        0.f), 49.f);
    float lty = fminf(fmaxf(fly,       0.f), 49.f);
    float rby = fminf(fmaxf(fly + 1.f, 0.f), 49.f);
    float pyc = fminf(fmaxf(py,        0.f), 49.f);
    float wxl = 1.f + (ltx - pxc);
    float wxr = 1.f - (rbx - pxc);
    float wyl = 1.f + (lty - pyc);
    float wyr = 1.f - (rby - pyc);
    int ix0 = (int)ltx, ix1 = (int)rbx, iy0 = (int)lty, iy1 = (int)rby;
    const float* xb = x + (long long)bc * (HIN * WIN);
    auto P = [&](int ix, int iy) -> float {
      if (ix < 1 || ix > HIN || iy < 1 || iy > WIN) return 0.f;  // zero pad
      return xb[(ix - 1) * WIN + (iy - 1)];
    };
    val = wxl * wyl * P(ix0, iy0) + wxr * wyr * P(ix1, iy1)
        + wxl * wyr * P(ix0, iy1) + wxr * wyl * P(ix1, iy0);
  }
  xo[idx] = (_Float16)val;
}

// ---------------- implicit-GEMM conv with v_wmma_f32_16x16x32_f16 ------------------
// Workgroup 256 thr = 8 waves in 4(M)x2(N); tile M=128 co x N=64 pix (8x8 spatial).
// A tile staged via async global->LDS b128 (ASYNCcnt); both fragment paths read
// LDS as contiguous 16B chunks (ds_load_b128).
__global__ __launch_bounds__(256) void conv_wmma_kernel(const _Float16* __restrict__ wh,
                                                        const _Float16* __restrict__ xo,
                                                        float* __restrict__ y) {
  __shared__ __align__(16) _Float16 As[128][40]; // 128 co rows x 32 K (+8 pad)
  __shared__ __align__(16) _Float16 Bs[64][40];  // 64 pix x 32 K (+8 pad), transposed

  const int tid  = threadIdx.x;
  const int lane = tid & 31;
  const int wv   = tid >> 5;
  const int wm   = wv & 3;       // wave M quadrant (32 rows each)
  const int wn   = wv >> 2;      // wave N half (32 cols each)
  const int lrow = lane & 15;
  const int hi   = lane >> 4;    // lane half

  const int coBase = blockIdx.x * 128;
  const int ow0    = blockIdx.y * 8;
  const int ohT    = blockIdx.z % 18;
  const int bb     = blockIdx.z / 18;
  const int oh0    = ohT * 8;

  const v8f vzero = {0.f,0.f,0.f,0.f,0.f,0.f,0.f,0.f};
  v8f acc[2][2];
  acc[0][0] = vzero; acc[0][1] = vzero; acc[1][0] = vzero; acc[1][1] = vzero;

  const int arow = tid >> 1;             // 0..127
  const int acol = (tid & 1) * 16;       // 0 or 16
  const int bkk  = tid >> 3;             // 0..31
  const int blh  = tid & 7;              // 0..7

  for (int k0 = 0; k0 < KTOT; k0 += 32) {
    // ---- stage A tile: 16 contiguous f16 (32 B, 16B-aligned) per thread ----
    const _Float16* ag = wh + (long long)(coBase + arow) * KTOT + k0 + acol;
    _Float16* al = &As[arow][acol];
#if USE_ASYNC
    __builtin_amdgcn_global_load_async_to_lds_b128(ASG(ag), ASL(al), 0, 0);
    __builtin_amdgcn_global_load_async_to_lds_b128(ASG(ag + 8), ASL(al + 8), 0, 0);
#else
    #pragma unroll
    for (int q = 0; q < 16; ++q) al[q] = ag[q];
#endif
    // ---- stage B tile (im2col row of 8 f16, 2B-aligned -> scalar path) ----
    {
      int k  = k0 + bkk;
      int ci = k / 9;
      int r  = k - ci * 9;
      int kh = r / 3;
      int kw = r - kh * 3;
      const _Float16* bg = xo +
        ((long long)(bb * C_IN + ci) * XH + (oh0 + blh + kh)) * XW + (ow0 + kw);
      #pragma unroll
      for (int q = 0; q < 8; ++q) Bs[blh * 8 + q][bkk] = bg[q];
    }
#if USE_ASYNC
    __builtin_amdgcn_s_wait_asynccnt(0);
#endif
    __syncthreads();

    v16h aF[2], bF[2];
    #pragma unroll
    for (int s = 0; s < 2; ++s) {
      // A fragment: lanes 0-15 hold K {0..7,16..23}, lanes 16-31 hold {8..15,24..31}
      int row = wm * 32 + s * 16 + lrow;
      int kb  = hi * 8;
      v8h alo = *(const v8h*)&As[row][kb];
      v8h ahi = *(const v8h*)&As[row][kb + 16];
      #pragma unroll
      for (int t = 0; t < 8; ++t) { aF[s][t] = alo[t]; aF[s][t + 8] = ahi[t]; }
      // B fragment: lanes 0-15 hold K {0..7,16..23}, lanes 16-31 hold {8..15,24..31}, col = lane%16
      int col = wn * 32 + s * 16 + lrow;
      v8h blo = *(const v8h*)&Bs[col][hi * 8];
      v8h bhi = *(const v8h*)&Bs[col][hi * 8 + 16];
      #pragma unroll
      for (int t = 0; t < 8; ++t) { bF[s][t] = blo[t]; bF[s][t + 8] = bhi[t]; }
    }
    #pragma unroll
    for (int i = 0; i < 2; ++i)
      #pragma unroll
      for (int j = 0; j < 2; ++j)
        acc[i][j] = __builtin_amdgcn_wmma_f32_16x16x32_f16(
            false, aF[i], false, bF[j], (short)0, acc[i][j], false, false);
    __syncthreads();
  }

  // store: C/D layout -> M = vgpr + 8*hi, N = lane%16
  #pragma unroll
  for (int i = 0; i < 2; ++i) {
    #pragma unroll
    for (int j = 0; j < 2; ++j) {
      int p  = wn * 32 + j * 16 + lrow;
      int lh = p >> 3, lw = p & 7;
      int oh = oh0 + lh, ow = ow0 + lw;
      if (oh < HOUT && ow < WOUT) {
        #pragma unroll
        for (int v = 0; v < 8; ++v) {
          int co = coBase + wm * 32 + i * 16 + hi * 8 + v;
          y[((long long)(bb * C_OUT + co) * HOUT + oh) * WOUT + ow] = acc[i][j][v];
        }
      }
    }
  }
}

// ---------------- per-(b,c) BN+ReLU sum over branches + spatial mean ---------------
__global__ __launch_bounds__(256) void reduce_kernel(const float* __restrict__ ys,
    const float* __restrict__ g, const float* __restrict__ bta,
    const float* __restrict__ mu, const float* __restrict__ var,
    float* __restrict__ fea_s) {
  int bc = blockIdx.x;          // b*256 + c
  int bb = bc >> 8, c = bc & 255;
  __shared__ float red[256];
  float scale[4], shift[4];
  #pragma unroll
  for (int m = 0; m < 4; ++m) {
    float s  = g[m * C_IN + c] * rsqrtf(var[m * C_IN + c] + BN_EPS);
    scale[m] = s;
    shift[m] = bta[m * C_IN + c] - mu[m * C_IN + c] * s;
  }
  const long long strideA = (long long)BATCH * C_OUT * NPIX;
  const long long base    = (long long)(bb * C_OUT + c) * NPIX;
  float sum = 0.f;
  for (int pix = threadIdx.x; pix < NPIX; pix += 256) {
    #pragma unroll
    for (int m = 0; m < 4; ++m) {
      float v = ys[(m + 1) * strideA + base + pix];
      sum += fmaxf(v * scale[m] + shift[m], 0.f);
    }
  }
  red[threadIdx.x] = sum;
  __syncthreads();
  for (int s = 128; s > 0; s >>= 1) {
    if (threadIdx.x < s) red[threadIdx.x] += red[threadIdx.x + s];
    __syncthreads();
  }
  if (threadIdx.x == 0) fea_s[bc] = red[0] / (float)NPIX;
}

// ---------------- fc1 -> fc2 -> softmax over branches ------------------------------
__global__ __launch_bounds__(256) void att_kernel(const float* __restrict__ fea_s,
    const float* __restrict__ fc1w, const float* __restrict__ fc1b,
    const float* __restrict__ fc2w, const float* __restrict__ fc2b,
    float* __restrict__ att) {
  int bb = blockIdx.x;
  __shared__ float z[32];
  int t = threadIdx.x;
  if (t < 32) {
    float s = fc1b[t];
    for (int c = 0; c < C_IN; ++c) s += fea_s[bb * C_IN + c] * fc1w[t * C_IN + c];
    z[t] = s;
  }
  __syncthreads();
  int c = t;
  float a[4];
  float mx = -1e30f;
  #pragma unroll
  for (int m = 0; m < 4; ++m) {
    float s = fc2b[m * C_IN + c];
    for (int d = 0; d < 32; ++d) s += z[d] * fc2w[(m * C_IN + c) * 32 + d];
    a[m] = s; mx = fmaxf(mx, s);
  }
  float den = 0.f;
  #pragma unroll
  for (int m = 0; m < 4; ++m) { a[m] = expf(a[m] - mx); den += a[m]; }
  #pragma unroll
  for (int m = 0; m < 4; ++m) att[(bb * 4 + m) * C_IN + c] = a[m] / den;
}

// ---------------- final weighted fusion + identity branch --------------------------
__global__ __launch_bounds__(256) void final_kernel(const float* __restrict__ ys,
    const float* __restrict__ g, const float* __restrict__ bta,
    const float* __restrict__ mu, const float* __restrict__ var,
    const float* __restrict__ att, float* __restrict__ out) {
  long long idx = (long long)blockIdx.x * 256 + threadIdx.x;
  const long long total = (long long)BATCH * C_OUT * NPIX;
  if (idx >= total) return;
  int bc = (int)(idx / NPIX);
  int bb = bc >> 8, c = bc & 255;
  float v = ys[idx];  // angle-0 branch
  #pragma unroll
  for (int m = 0; m < 4; ++m) {
    float s  = g[m * C_IN + c] * rsqrtf(var[m * C_IN + c] + BN_EPS);
    float sh = bta[m * C_IN + c] - mu[m * C_IN + c] * s;
    float ym = ys[(m + 1) * total + idx];
    v += fmaxf(ym * s + sh, 0.f) * att[(bb * 4 + m) * C_IN + c];
  }
  out[idx] = v;
}

extern "C" void kernel_launch(void* const* d_in, const int* in_sizes, int n_in,
                              void* d_out, int out_size, void* d_ws, size_t ws_size,
                              hipStream_t stream) {
  const float* x      = (const float*)d_in[0];
  const float* conv_w = (const float*)d_in[1];
  const float* bn_g   = (const float*)d_in[2];
  const float* bn_b   = (const float*)d_in[3];
  const float* bn_m   = (const float*)d_in[4];
  const float* bn_v   = (const float*)d_in[5];
  const float* fc1w   = (const float*)d_in[6];
  const float* fc1b   = (const float*)d_in[7];
  const float* fc2w   = (const float*)d_in[8];
  const float* fc2b   = (const float*)d_in[9];
  float* out = (float*)d_out;

  char* ws = (char*)d_ws;
  size_t off = 0;
  auto carve = [&](size_t bytes) -> void* {
    void* p = ws + off;
    off = (off + bytes + 255) & ~(size_t)255;
    return p;
  };
  const size_t nW   = (size_t)C_OUT * KTOT;                 // 589824
  const size_t nXO  = (size_t)BATCH * C_IN * XH * XW;       // 11,960,320
  const size_t nY1  = (size_t)BATCH * C_OUT * NPIX;         // 10,323,968
  _Float16* wh    = (_Float16*)carve(nW * sizeof(_Float16));
  _Float16* xo    = (_Float16*)carve(nXO * sizeof(_Float16));
  float*    ys    = (float*)carve(5 * nY1 * sizeof(float)); // ~206 MB
  float*    fea_s = (float*)carve((size_t)BATCH * C_IN * sizeof(float));
  float*    att   = (float*)carve((size_t)BATCH * 4 * C_IN * sizeof(float));

  convert_w_kernel<<<(int)(nW / 256), 256, 0, stream>>>(conv_w, wh, (int)nW);

  dim3 cgrid(2, 18, 2 * 18); // (co tiles of 128, ow tiles of 8, b * oh tiles of 8)
  for (int a = 0; a < 5; ++a) {
    sample_kernel<<<(int)((nXO + 255) / 256), 256, 0, stream>>>(x, xo, a);
    conv_wmma_kernel<<<cgrid, 256, 0, stream>>>(wh, xo, ys + (size_t)a * nY1);
  }

  reduce_kernel<<<BATCH * C_IN, 256, 0, stream>>>(ys, bn_g, bn_b, bn_m, bn_v, fea_s);
  att_kernel<<<BATCH, 256, 0, stream>>>(fea_s, fc1w, fc1b, fc2w, fc2b, att);
  final_kernel<<<(int)((nY1 + 255) / 256), 256, 0, stream>>>(
      ys, bn_g, bn_b, bn_m, bn_v, att, out);
}
